// spiking_coESN_27925877358642
// MI455X (gfx1250) — matrix operand order repaired and checked by reading, not compiled, not run on hardware
//
#include <hip/hip_runtime.h>
#include <cmath>

#define NH      2048
#define BATCH   128
#define LSTEPS  128
#define DTC     0.042f
#define THETA   0.1f
#define ALPHAc  0.3f
#define BETAc   0.4f

typedef __attribute__((ext_vector_type(16))) __bf16 v16bf;
typedef __attribute__((ext_vector_type(8)))  __bf16 v8bf;
typedef __attribute__((ext_vector_type(8)))  float  v8f;

union AFrag { v16bf v; v8bf h[2]; };

// branchless tanh: 1 - 2/(e^{2x}+1), via v_exp_f32 + v_rcp_f32 (no EXEC divergence)
__device__ __forceinline__ float fast_tanh(float v) {
  float e = __builtin_amdgcn_exp2f(v * 2.885390081777927f);  // e^{2v}
  return 1.0f - 2.0f * __builtin_amdgcn_rcpf(e + 1.0f);
}

// ---------------- init: zero all state ----------------
__global__ __launch_bounds__(256) void esn_init(float* __restrict__ hy,
                                                __bf16* __restrict__ hyb,
                                                float* __restrict__ hz,
                                                float* __restrict__ rf,
                                                float* __restrict__ spk) {
  int i = blockIdx.x * 256 + threadIdx.x;
  if (i < BATCH * NH) {
    hy[i]  = 0.f;
    hyb[i] = (__bf16)0.f;
    hz[i]  = 0.f;
    rf[i]  = 0.f;
    spk[i] = 0.f;
  }
}

// ------- one-time: h2h (K x N, f32) -> h2h^T (N x K, bf16) -------
__global__ __launch_bounds__(256) void esn_cvt(const float* __restrict__ h2h,
                                               __bf16* __restrict__ h2hT) {
  int i = blockIdx.x * 256 + threadIdx.x;   // i = n*NH + k  (coalesced write)
  if (i < NH * NH) {
    int n = i >> 11;
    int k = i & (NH - 1);
    h2hT[i] = (__bf16)h2h[k * NH + n];
  }
}

// ---------------- one recurrence step ----------------
// grid = 64 blocks x 128 threads (4 waves); each wave owns a 32x32 C macro-tile
// (2 M-tiles x 2 N-tiles -> 4 independent WMMA accumulator chains).
__global__ __launch_bounds__(128) void esn_step(
    const float*  __restrict__ x,        // (B, L)  (N_INP==1)
    const __bf16* __restrict__ h2hT,     // (N, K) bf16
    const float*  __restrict__ x2h,      // (N)
    const float*  __restrict__ bias,     // (N)
    const float*  __restrict__ gamma,    // (N)
    const float*  __restrict__ eps,      // (N)
    const float*  __restrict__ hy_old,   // (B, N) f32
    const __bf16* __restrict__ hy_old_bf,// (B, N) bf16 shadow
    float*        __restrict__ hy_new,
    __bf16*       __restrict__ hy_new_bf,
    float*        __restrict__ hz,
    float*        __restrict__ rf,
    float*        __restrict__ spk,
    int t, float ref_decay)
{
  const int lane = threadIdx.x & 31;
  const int wave = threadIdx.x >> 5;
  const int mb   = blockIdx.x >> 4;                  // 0..3   (32-row macro block)
  const int njob = ((blockIdx.x & 15) << 2) + wave;  // 0..63  (32-col macro block)
  const int hi   = lane >> 4;                        // half-wave
  const int l15  = lane & 15;

  // A fragments: rows M = mb*32 + l15 (+16); 16 contiguous K per lane at +hi*16
  const __bf16* aRow0 = hy_old_bf + (size_t)(mb * 32 + l15) * NH + hi * 16;
  const __bf16* aRow1 = aRow0 + (size_t)16 * NH;
  // B fragments: cols N = njob*32 + l15 (+16); 16 contiguous K per lane at +hi*16
  const __bf16* bRow0 = h2hT + (size_t)(njob * 32 + l15) * NH + hi * 16;
  const __bf16* bRow1 = bRow0 + (size_t)16 * NH;

  v8f acc00 = {}, acc01 = {}, acc10 = {}, acc11 = {};

  for (int kc = 0; kc < NH; kc += 32) {
    AFrag a0, a1;
    a0.h[0] = *(const v8bf*)(aRow0 + kc);
    a0.h[1] = *(const v8bf*)(aRow0 + kc + 8);
    a1.h[0] = *(const v8bf*)(aRow1 + kc);
    a1.h[1] = *(const v8bf*)(aRow1 + kc + 8);
    v16bf b0 = *(const v16bf*)(bRow0 + kc);
    v16bf b1 = *(const v16bf*)(bRow1 + kc);

    // pull next K-chunk toward the WGP (global_prefetch_b8)
    __builtin_prefetch(bRow0 + kc + 32, 0, 3);
    __builtin_prefetch(bRow1 + kc + 32, 0, 3);
    __builtin_prefetch(aRow0 + kc + 32, 0, 3);

    acc00 = __builtin_amdgcn_wmma_f32_16x16x32_bf16(false, a0.v, false, b0,
                                                    (short)0, acc00, false, false);
    acc01 = __builtin_amdgcn_wmma_f32_16x16x32_bf16(false, a0.v, false, b1,
                                                    (short)0, acc01, false, false);
    acc10 = __builtin_amdgcn_wmma_f32_16x16x32_bf16(false, a1.v, false, b0,
                                                    (short)0, acc10, false, false);
    acc11 = __builtin_amdgcn_wmma_f32_16x16x32_bf16(false, a1.v, false, b1,
                                                    (short)0, acc11, false, false);
  }

  // pointwise epilogue: 4 tiles; tile (ti,tj): M = mb*32+ti*16+j+hi*8, N = njob*32+tj*16+l15
  #pragma unroll
  for (int tj = 0; tj < 2; ++tj) {
    const int n = njob * 32 + tj * 16 + l15;
    const float x2hn = x2h[n];
    const float bn   = bias[n];
    const float gn   = gamma[n];
    const float en   = eps[n];
    #pragma unroll
    for (int ti = 0; ti < 2; ++ti) {
      const v8f acc = (ti == 0) ? (tj == 0 ? acc00 : acc01)
                                : (tj == 0 ? acc10 : acc11);
      #pragma unroll
      for (int j = 0; j < 8; ++j) {
        const int    b   = mb * 32 + ti * 16 + j + hi * 8;
        const size_t idx = (size_t)b * NH + n;
        const float pre  = fast_tanh(x[b * LSTEPS + t] * x2hn + acc[j] + bn);
        const float hyo  = hy_old[idx];
        float hzv = hz[idx];
        hzv = hzv + DTC * (pre - gn * hyo - en * hzv);
        float hyv = hyo + DTC * hzv;
        float r   = rf[idx];
        const float s = (hyv - THETA - r > 0.f) ? 1.f : 0.f;  // v_cmp + v_cndmask
        hyv *= (1.f - s * ALPHAc);
        hzv *= (1.f - s * BETAc);
        r    = r * ref_decay + s;
        hy_new[idx]    = hyv;
        hy_new_bf[idx] = (__bf16)hyv;
        hz[idx]  = hzv;
        rf[idx]  = r;
        spk[idx] += s;
      }
    }
  }
}

// ---------------- finalize: mean spike rate ----------------
__global__ __launch_bounds__(256) void esn_fin(const float* __restrict__ spk,
                                               float* __restrict__ out) {
  int i = blockIdx.x * 256 + threadIdx.x;
  if (i < BATCH * NH) out[i] = spk[i] * (1.0f / (float)LSTEPS);
}

extern "C" void kernel_launch(void* const* d_in, const int* in_sizes, int n_in,
                              void* d_out, int out_size, void* d_ws, size_t ws_size,
                              hipStream_t stream) {
  const float* x     = (const float*)d_in[0];  // (128,128,1)
  const float* h2h   = (const float*)d_in[1];  // (2048,2048)
  const float* x2h   = (const float*)d_in[2];  // (2048)
  const float* bias  = (const float*)d_in[3];
  const float* gamma = (const float*)d_in[4];
  const float* eps   = (const float*)d_in[5];

  char* p = (char*)d_ws;
  __bf16* h2hT  = (__bf16*)p;  p += (size_t)NH * NH * sizeof(__bf16);       // 8 MB
  float*  hyA   = (float*)p;   p += (size_t)BATCH * NH * sizeof(float);     // 1 MB
  float*  hyB   = (float*)p;   p += (size_t)BATCH * NH * sizeof(float);
  __bf16* hyAbf = (__bf16*)p;  p += (size_t)BATCH * NH * sizeof(__bf16);
  __bf16* hyBbf = (__bf16*)p;  p += (size_t)BATCH * NH * sizeof(__bf16);
  float*  hz    = (float*)p;   p += (size_t)BATCH * NH * sizeof(float);
  float*  rf    = (float*)p;   p += (size_t)BATCH * NH * sizeof(float);
  float*  spk   = (float*)p;   p += (size_t)BATCH * NH * sizeof(float);

  const float ref_decay = expf(-DTC / 0.25f);

  const int elemsState = BATCH * NH;            // 262144
  const int elemsH2H   = NH * NH;               // 4194304

  esn_init<<<(elemsState + 255) / 256, 256, 0, stream>>>(hyA, hyAbf, hz, rf, spk);
  esn_cvt <<<(elemsH2H   + 255) / 256, 256, 0, stream>>>(h2h, h2hT);

  for (int t = 0; t < LSTEPS; ++t) {
    const float*  hyo  = (t & 1) ? hyB   : hyA;
    const __bf16* hyob = (t & 1) ? hyBbf : hyAbf;
    float*        hyn  = (t & 1) ? hyA   : hyB;
    __bf16*       hynb = (t & 1) ? hyAbf : hyBbf;
    esn_step<<<64, 128, 0, stream>>>(x, h2hT, x2h, bias, gamma, eps,
                                     hyo, hyob, hyn, hynb, hz, rf, spk,
                                     t, ref_decay);
  }

  esn_fin<<<(elemsState + 255) / 256, 256, 0, stream>>>(spk, (float*)d_out);
}
